// DEQLayer_54649163874707
// MI455X (gfx1250) — matrix operand order repaired and checked by reading, not compiled, not run on hardware
//
#include <hip/hip_runtime.h>
#include <math.h>

typedef __attribute__((ext_vector_type(16))) _Float16 v16h;
typedef __attribute__((ext_vector_type(8)))  _Float16 v8h;
typedef __attribute__((ext_vector_type(8)))  float    v8f;

#define B_TOTAL 2048
#define X_DIM   256
#define LATENT  256
#define HIDDEN  1024
#define IN_DIM  512
#define HIST    6
#define REG_EPS 1e-4f

__device__ __forceinline__ float dev_tanh(float x) {
#if __has_builtin(__builtin_amdgcn_tanhf)
    return __builtin_amdgcn_tanhf(x);   // CDNA5 V_TANH_F32 (TRANS pipe)
#else
    return tanhf(x);
#endif
}

__device__ __forceinline__ void wait_async0() {
#if __has_builtin(__builtin_amdgcn_s_wait_asynccnt)
    __builtin_amdgcn_s_wait_asynccnt(0);
#else
    asm volatile("s_wait_asynccnt 0" ::: "memory");
#endif
}

// ---------------------------------------------------------------------------
// Spectral norm sigma via power iteration (mirrors reference exactly).
// One workgroup per matrix; W is rows x cols row-major. Writes sigma scalar.
// ---------------------------------------------------------------------------
__global__ __launch_bounds__(256) void power_iter_kernel(
    const float* __restrict__ W, int rows, int cols, float* __restrict__ sigma_out)
{
    __shared__ float sh_u[1024];
    __shared__ float sh_v[1024];
    __shared__ float sh_red[256];
    const int t = threadIdx.x;
    const int NT = 256;

    float u0 = 1.0f / (sqrtf((float)rows) + 1e-12f);   // l2norm(ones)
    for (int i = t; i < rows; i += NT) sh_u[i] = u0;
    __syncthreads();

    for (int it = 0; it < 20; ++it) {
        // v = l2norm(W^T u)
        for (int j = t; j < cols; j += NT) {
            float s = 0.f;
            for (int i = 0; i < rows; ++i) s += W[(size_t)i * cols + j] * sh_u[i];
            sh_v[j] = s;
        }
        __syncthreads();
        float ls = 0.f;
        for (int j = t; j < cols; j += NT) ls += sh_v[j] * sh_v[j];
        sh_red[t] = ls; __syncthreads();
        for (int o = 128; o > 0; o >>= 1) { if (t < o) sh_red[t] += sh_red[t + o]; __syncthreads(); }
        float inv = 1.0f / (sqrtf(sh_red[0]) + 1e-12f);
        __syncthreads();
        for (int j = t; j < cols; j += NT) sh_v[j] *= inv;
        __syncthreads();
        // u = l2norm(W v)
        for (int i = t; i < rows; i += NT) {
            float s = 0.f;
            for (int j = 0; j < cols; ++j) s += W[(size_t)i * cols + j] * sh_v[j];
            sh_u[i] = s;
        }
        __syncthreads();
        ls = 0.f;
        for (int i = t; i < rows; i += NT) ls += sh_u[i] * sh_u[i];
        sh_red[t] = ls; __syncthreads();
        for (int o = 128; o > 0; o >>= 1) { if (t < o) sh_red[t] += sh_red[t + o]; __syncthreads(); }
        inv = 1.0f / (sqrtf(sh_red[0]) + 1e-12f);
        __syncthreads();
        for (int i = t; i < rows; i += NT) sh_u[i] *= inv;
        __syncthreads();
    }
    // final v = l2norm(W^T u)
    for (int j = t; j < cols; j += NT) {
        float s = 0.f;
        for (int i = 0; i < rows; ++i) s += W[(size_t)i * cols + j] * sh_u[i];
        sh_v[j] = s;
    }
    __syncthreads();
    float ls = 0.f;
    for (int j = t; j < cols; j += NT) ls += sh_v[j] * sh_v[j];
    sh_red[t] = ls; __syncthreads();
    for (int o = 128; o > 0; o >>= 1) { if (t < o) sh_red[t] += sh_red[t + o]; __syncthreads(); }
    float inv = 1.0f / (sqrtf(sh_red[0]) + 1e-12f);
    __syncthreads();
    for (int j = t; j < cols; j += NT) sh_v[j] *= inv;
    __syncthreads();
    // sigma = u . (W v)
    ls = 0.f;
    for (int i = t; i < rows; i += NT) {
        float s = 0.f;
        for (int j = 0; j < cols; ++j) s += W[(size_t)i * cols + j] * sh_v[j];
        ls += sh_u[i] * s;
    }
    sh_red[t] = ls; __syncthreads();
    for (int o = 128; o > 0; o >>= 1) { if (t < o) sh_red[t] += sh_red[t + o]; __syncthreads(); }
    if (t == 0) sigma_out[0] = sh_red[0];
}

// W_f16 = f16(W / sigma)
__global__ void scale_cvt_kernel(const float* __restrict__ W, const float* __restrict__ sigma,
                                 _Float16* __restrict__ out, int n)
{
    int i = blockIdx.x * blockDim.x + threadIdx.x;
    if (i < n) out[i] = (_Float16)(W[i] / sigma[0]);
}

// in_f16[b, 0:256] = 0 (z0), in_f16[b, 256:512] = f16(context[b])
__global__ void init_in_kernel(const float* __restrict__ ctx, _Float16* __restrict__ inh)
{
    int i = blockIdx.x * blockDim.x + threadIdx.x;
    if (i >= B_TOTAL * IN_DIM) return;
    int b = i >> 9, c = i & 511;
    inh[i] = (c < LATENT) ? (_Float16)0.0f : (_Float16)ctx[(size_t)b * X_DIM + (c - LATENT)];
}

__global__ void zero_f32_kernel(float* __restrict__ p, int n)
{
    int i = blockIdx.x * blockDim.x + threadIdx.x;
    if (i < n) p[i] = 0.0f;
}

// ---------------------------------------------------------------------------
// Fused f(z,ctx): one block = 16 batch rows.
//   GEMM1 (K=512 -> 1024 cols) via v_wmma_f32_16x16x32_f16, bias+tanh -> LDS f16
//   GEMM2 (K=1024 -> 256 cols) via WMMA, bias+tanh -> F slot (f32) + in_f16 z-part
// B operands are batch-preloaded into registers so the 16 global b128 loads of
// a k-step stream as one clause and WMMAs don't each stall on loadcnt==0.
// ---------------------------------------------------------------------------
__global__ __launch_bounds__(256) void deq_eval_kernel(
    const _Float16* __restrict__ inh,   // [2048,512] f16 (z | ctx)
    const _Float16* __restrict__ W1h,   // [1024,512] f16 (spectral-normalized)
    const float*    __restrict__ b1,    // [1024]
    const _Float16* __restrict__ W2h,   // [256,1024] f16
    const float*    __restrict__ b2,    // [256]
    float*          __restrict__ Fout,  // [2048,256] f32
    _Float16*       __restrict__ zh)    // write f16(z) into [2048,512] cols 0..255
{
    __shared__ __align__(16) _Float16 lds_in[16 * IN_DIM];   // 16 KB
    __shared__ __align__(16) _Float16 lds_h [16 * HIDDEN];   // 32 KB
    const int tid  = threadIdx.x;
    const int wave = tid >> 5;
    const int lane = tid & 31;
    const int b0   = blockIdx.x * 16;
    const int nr   = lane & 15;     // N (and A-row M) selector
    const int koff = lane >> 4;     // lane-group K offset selector

    // stage the 16x512 input tile (contiguous 16KB) via async Global->LDS DMA
    // (ASYNCcnt path) when available; VALU copy fallback otherwise.
#if __has_builtin(__builtin_amdgcn_global_load_async_to_lds_b128)
    {
        // exact pointee type from the builtin's prototype: int __vector_size__(16)
        typedef int i4vec __attribute__((__vector_size__(16)));
        typedef __attribute__((address_space(1))) i4vec g_i4;   // global
        typedef __attribute__((address_space(3))) i4vec l_i4;   // LDS
        const _Float16* src = inh + (size_t)b0 * IN_DIM + (size_t)tid * 8;  // 16B per lane
        _Float16*       dst = lds_in + (size_t)tid * 8;
        #pragma unroll
        for (int c = 0; c < 4; ++c) {   // 4 sweeps x 256 lanes x 16B = 16 KB
            __builtin_amdgcn_global_load_async_to_lds_b128(
                (g_i4*)(uintptr_t)(src + (size_t)c * 256 * 8),
                (l_i4*)(uint32_t)(uintptr_t)(dst + (size_t)c * 256 * 8),
                0, 0);
        }
        wait_async0();
    }
#else
    {
        const uint4* src = (const uint4*)(inh + (size_t)b0 * IN_DIM);
        uint4* dst = (uint4*)lds_in;
        #pragma unroll
        for (int c = tid; c < 16 * IN_DIM / 8; c += 256) dst[c] = src[c];
    }
#endif
    __syncthreads();

    // ---- GEMM1: each wave owns 8 of the 64 N-tiles ----
    {
        v8f acc[8] = {};
        const int nbase = wave * 8;
        const _Float16* wbase = W1h + (size_t)(nbase * 16 + nr) * IN_DIM + koff * 8;
        for (int kt = 0; kt < IN_DIM / 32; ++kt) {
            const int k0 = kt * 32;
            // A fragment (16x32 f16): lanes 0-15 K=0..7,16..23 ; lanes 16-31 K=8..15,24..31
            v16h a;
            {
                const v8h* p0 = (const v8h*)&lds_in[nr * IN_DIM + k0 + koff * 8];
                const v8h* p1 = (const v8h*)&lds_in[nr * IN_DIM + k0 + 16 + koff * 8];
                v8h lo = *p0, hi = *p1;
                #pragma unroll
                for (int e = 0; e < 8; ++e) { a[e] = lo[e]; a[8 + e] = hi[e]; }
            }
            // batch-preload all 8 B fragments (16 b128 loads -> one clause)
            v16h bfrag[8];
            #pragma unroll
            for (int n = 0; n < 8; ++n) {
                const v8h* q = (const v8h*)(wbase + (size_t)n * 16 * IN_DIM + k0);
                v8h blo = q[0], bhi = q[2];
                #pragma unroll
                for (int e = 0; e < 8; ++e) { bfrag[n][e] = blo[e]; bfrag[n][8 + e] = bhi[e]; }
            }
            #pragma unroll
            for (int n = 0; n < 8; ++n)
                acc[n] = __builtin_amdgcn_wmma_f32_16x16x32_f16(
                    false, a, false, bfrag[n], (short)0, acc[n], false, false);
        }
        // bias + tanh, store h as f16 row-major in LDS
        #pragma unroll
        for (int n = 0; n < 8; ++n) {
            const int col = (nbase + n) * 16 + nr;
            const float bias = b1[col];
            #pragma unroll
            for (int r = 0; r < 8; ++r) {
                const int M = r + 8 * koff;     // C/D layout: M = vgpr + 8*(lane>=16)
                lds_h[M * HIDDEN + col] = (_Float16)dev_tanh(acc[n][r] + bias);
            }
        }
    }
    __syncthreads();

    // ---- GEMM2: each wave owns 2 of the 16 N-tiles ----
    {
        v8f acc[2] = {};
        const int nbase = wave * 2;
        const _Float16* wbase = W2h + (size_t)(nbase * 16 + nr) * HIDDEN + koff * 8;
        for (int kt = 0; kt < HIDDEN / 32; ++kt) {
            const int k0 = kt * 32;
            v16h a;
            {
                const v8h* p0 = (const v8h*)&lds_h[nr * HIDDEN + k0 + koff * 8];
                const v8h* p1 = (const v8h*)&lds_h[nr * HIDDEN + k0 + 16 + koff * 8];
                v8h lo = *p0, hi = *p1;
                #pragma unroll
                for (int e = 0; e < 8; ++e) { a[e] = lo[e]; a[8 + e] = hi[e]; }
            }
            v16h bfrag[2];
            #pragma unroll
            for (int n = 0; n < 2; ++n) {
                const v8h* q = (const v8h*)(wbase + (size_t)n * 16 * HIDDEN + k0);
                v8h blo = q[0], bhi = q[2];
                #pragma unroll
                for (int e = 0; e < 8; ++e) { bfrag[n][e] = blo[e]; bfrag[n][8 + e] = bhi[e]; }
            }
            #pragma unroll
            for (int n = 0; n < 2; ++n)
                acc[n] = __builtin_amdgcn_wmma_f32_16x16x32_f16(
                    false, a, false, bfrag[n], (short)0, acc[n], false, false);
        }
        #pragma unroll
        for (int n = 0; n < 2; ++n) {
            const int col = (nbase + n) * 16 + nr;
            const float bias = b2[col];
            #pragma unroll
            for (int r = 0; r < 8; ++r) {
                const int M = r + 8 * koff;
                const int row = b0 + M;
                const float z = dev_tanh(acc[n][r] + bias);
                Fout[(size_t)row * LATENT + col] = z;
                zh[(size_t)row * IN_DIM + col]   = (_Float16)z;
            }
        }
    }
}

// ---------------------------------------------------------------------------
// Anderson update at loop index k (k=2..24). One wave per batch row.
// Reads histories X_{k-mk..k}, F_{k-mk..k} from ring slots (g % 6), builds the
// regularized normal equations (mk<=5), Gauss-Jordan solve (lane-redundant),
// writes x_{k+1} (f32) and f16(x) into the input staging buffer.
// Invalid columns (i>mk) are masked to zero -> alpha_i = Bv_i/REG = 0.
// ---------------------------------------------------------------------------
__global__ __launch_bounds__(256) void anderson_step_kernel(
    const float* __restrict__ Xh, const float* __restrict__ Fh, int k,
    float* __restrict__ x_out, _Float16* __restrict__ inh)
{
    const int wave = threadIdx.x >> 5;
    const int lane = threadIdx.x & 31;
    const int b = blockIdx.x * 8 + wave;
    const int mk = (k < 5) ? k : 5;
    const size_t slab = (size_t)B_TOTAL * LATENT;

    float sA[5][5] = {};
    float sB[5]    = {};

    #pragma unroll 1
    for (int dd = 0; dd < LATENT / 32; ++dd) {
        const int d = lane + dd * 32;
        float Xv[6], Fv[6];
        #pragma unroll
        for (int j = 0; j < 6; ++j) {            // global index g = k - j
            int g = k - j; if (g < 0) g = 0;     // clamped slots are valid+finite; masked below
            const size_t off = (size_t)(g % HIST) * slab + (size_t)b * LATENT + d;
            Xv[j] = Xh[off];
            Fv[j] = Fh[off];
        }
        const float r = Fv[0] - Xv[0];
        float dg[5];
        #pragma unroll
        for (int i = 1; i <= 5; ++i) {
            const float m  = (i <= mk) ? 1.0f : 0.0f;
            const float df = m * (Fv[i - 1] - Fv[i]);
            const float dx = m * (Xv[i - 1] - Xv[i]);
            dg[i - 1] = df - dx;
        }
        #pragma unroll
        for (int i = 0; i < 5; ++i) {
            sB[i] += dg[i] * r;
            #pragma unroll
            for (int j = i; j < 5; ++j) sA[i][j] += dg[i] * dg[j];
        }
    }

    // wave butterfly reductions
    #pragma unroll
    for (int i = 0; i < 5; ++i) {
        #pragma unroll
        for (int o = 16; o > 0; o >>= 1) sB[i] += __shfl_xor(sB[i], o, 32);
        #pragma unroll
        for (int j = i; j < 5; ++j)
            #pragma unroll
            for (int o = 16; o > 0; o >>= 1) sA[i][j] += __shfl_xor(sA[i][j], o, 32);
    }

    // 5x5 Gauss-Jordan (SPD + REG*I, no pivoting needed); lane-redundant
    float M[5][5], rhs[5];
    #pragma unroll
    for (int i = 0; i < 5; ++i) {
        rhs[i] = sB[i];
        #pragma unroll
        for (int j = 0; j < 5; ++j)
            M[i][j] = ((i <= j) ? sA[i][j] : sA[j][i]) + ((i == j) ? REG_EPS : 0.0f);
    }
    #pragma unroll
    for (int p = 0; p < 5; ++p) {
        const float inv = 1.0f / M[p][p];
        #pragma unroll
        for (int j = 0; j < 5; ++j) M[p][j] *= inv;
        rhs[p] *= inv;
        #pragma unroll
        for (int i = 0; i < 5; ++i) {
            if (i == p) continue;
            const float f = M[i][p];
            #pragma unroll
            for (int j = 0; j < 5; ++j) M[i][j] -= f * M[p][j];
            rhs[i] -= f * rhs[p];
        }
    }

    // x = F_k - dF @ alpha ; write f32 and f16 staging
    float*     xrow = x_out + (size_t)b * LATENT;
    _Float16*  zrow = inh   + (size_t)b * IN_DIM;
    #pragma unroll 1
    for (int dd = 0; dd < LATENT / 32; ++dd) {
        const int d = lane + dd * 32;
        float Fv[6];
        #pragma unroll
        for (int j = 0; j < 6; ++j) {
            int g = k - j; if (g < 0) g = 0;
            Fv[j] = Fh[(size_t)(g % HIST) * slab + (size_t)b * LATENT + d];
        }
        float x = Fv[0];
        #pragma unroll
        for (int i = 1; i <= 5; ++i) {
            const float m = (i <= mk) ? 1.0f : 0.0f;
            x -= rhs[i - 1] * (m * (Fv[i - 1] - Fv[i]));
        }
        xrow[d] = x;
        zrow[d] = (_Float16)x;
    }
}

// ---------------------------------------------------------------------------
extern "C" void kernel_launch(void* const* d_in, const int* in_sizes, int n_in,
                              void* d_out, int out_size, void* d_ws, size_t ws_size,
                              hipStream_t stream)
{
    const float* ctx = (const float*)d_in[0];   // [2048,256]
    const float* W1  = (const float*)d_in[1];   // [1024,512]
    const float* b1  = (const float*)d_in[2];   // [1024]
    const float* W2  = (const float*)d_in[3];   // [256,1024]
    const float* b2  = (const float*)d_in[4];   // [256]

    // workspace carve-up (~27.5 MB, all 16B-aligned)
    char* w = (char*)d_ws;
    float*     sig = (float*)w;                                   // 2 floats
    _Float16*  W1h = (_Float16*)(w + 64);                         // 1 MB
    _Float16*  W2h = (_Float16*)(w + 64 + 1048576);               // 512 KB
    _Float16*  inh = (_Float16*)(w + 64 + 1048576 + 524288);      // 2 MB
    float*     Xh  = (float*)(w + 64 + 1048576 + 524288 + 2097152);
    const size_t slab = (size_t)B_TOTAL * LATENT;                 // floats
    float*     Fh  = Xh + (size_t)HIST * slab;

    auto Xslot = [&](int g) { return Xh + (size_t)(g % HIST) * slab; };
    auto Fslot = [&](int g) { return Fh + (size_t)(g % HIST) * slab; };

    // 1) spectral norm sigmas + f16 normalized weights
    power_iter_kernel<<<1, 256, 0, stream>>>(W1, HIDDEN, IN_DIM, sig + 0);
    power_iter_kernel<<<1, 256, 0, stream>>>(W2, LATENT, HIDDEN, sig + 1);
    scale_cvt_kernel<<<(HIDDEN * IN_DIM + 255) / 256, 256, 0, stream>>>(W1, sig + 0, W1h, HIDDEN * IN_DIM);
    scale_cvt_kernel<<<(LATENT * HIDDEN + 255) / 256, 256, 0, stream>>>(W2, sig + 1, W2h, LATENT * HIDDEN);

    // 2) z0 = 0, staged input = [0 | ctx], X_0 = 0
    init_in_kernel<<<(B_TOTAL * IN_DIM + 255) / 256, 256, 0, stream>>>(ctx, inh);
    zero_f32_kernel<<<(int)((slab + 255) / 256), 256, 0, stream>>>(Xslot(0), (int)slab);

    // 3) warm-up (mirrors reference, incl. its duplicated step via copies)
    deq_eval_kernel<<<B_TOTAL / 16, 256, 0, stream>>>(inh, W1h, b1, W2h, b2, Fslot(0), inh); // F_0 = f(x0)
    (void)hipMemcpyAsync(Xslot(1), Fslot(0), slab * sizeof(float), hipMemcpyDeviceToDevice, stream); // X_1 = F_0
    deq_eval_kernel<<<B_TOTAL / 16, 256, 0, stream>>>(inh, W1h, b1, W2h, b2, Fslot(1), inh); // F_1 = f(F_0)
    (void)hipMemcpyAsync(Xslot(2), Fslot(0), slab * sizeof(float), hipMemcpyDeviceToDevice, stream); // X_2 = F_0
    (void)hipMemcpyAsync(Fslot(2), Fslot(1), slab * sizeof(float), hipMemcpyDeviceToDevice, stream); // F_2 = f(F_0) = F_1

    // 4) Anderson loop k = 2..24; final x (k=24) is the answer (X[-1]); its eval is dead.
    for (int k = 2; k <= 24; ++k) {
        float* xdst = (k == 24) ? (float*)d_out : Xslot(k + 1);
        anderson_step_kernel<<<B_TOTAL / 8, 256, 0, stream>>>(Xh, Fh, k, xdst, inh);
        if (k < 24)
            deq_eval_kernel<<<B_TOTAL / 16, 256, 0, stream>>>(inh, W1h, b1, W2h, b2, Fslot(k + 1), inh);
    }
}
